// HeteroGNN_42116449304614
// MI455X (gfx1250) — compile-verified, matrix operand order checked by reading, not run on hardware
//
#include <hip/hip_runtime.h>
#include <cstddef>
#include <cstdint>

// ---------------------------------------------------------------------------
// HeteroGNN (2-layer SAGE) forward for MI455X / gfx1250, wave32.
// Dense GEMMs via v_wmma_f32_16x16x32_f16 (f16 inputs, f32 accumulate).
// Irregular aggregation via global_atomic_add_f32 in fp32.
// A-tile staging via global_load_async_to_lds_b128 (ASYNCcnt path).
// ---------------------------------------------------------------------------

typedef __attribute__((ext_vector_type(16))) _Float16 v16h;
typedef __attribute__((ext_vector_type(8)))  _Float16 v8h;
typedef __attribute__((ext_vector_type(8)))  float    v8f;

#define TB 128  // 4 waves per block

// ---- WMMA fragment helpers -------------------------------------------------
// Assumed gfx1250 16-bit fragment lane mapping (ISA 7.12.2):
//   A (16x32, M x K): lane l (l<16) holds row M=l, halves: K = k0+0..7 then k0+16..23
//                     lane l+16 holds row M=l,       K = k0+8..15 then k0+24..31
//   B (32x16, K x N): symmetric: lane n holds col N=n with the same K split.
//   C/D (16x16)     : element r of lane l -> row M = r + (l<16?0:8), col N = l&15.

__device__ inline v16h cat8(v8h lo, v8h hi) {
  v16h r;
#pragma unroll
  for (int j = 0; j < 8; ++j) { r[j] = lo[j]; r[j + 8] = hi[j]; }
  return r;
}

__device__ inline v16h load_a_frag(const _Float16* sh, int m, int kbase, int K, int hsel) {
  const _Float16* p0 = sh + m * K + kbase + hsel;       // 16B aligned (hsel in {0,8})
  const _Float16* p1 = sh + m * K + kbase + 16 + hsel;
  v8h lo = *(const v8h*)p0;
  v8h hi = *(const v8h*)p1;
  return cat8(lo, hi);
}

// Packed B: [(nt*nKt + kt)*32 + lane] * 16 halfs, fragment-element order.
__device__ inline v16h load_b_frag(const _Float16* __restrict__ Bpk,
                                   int nt, int kt, int nKt, int lane) {
  const _Float16* p = Bpk + ((((size_t)nt * nKt + kt) * 32 + lane) << 4);
  v8h lo = *(const v8h*)p;
  v8h hi = *(const v8h*)(p + 8);
  return cat8(lo, hi);
}

__device__ inline v8f wmma_f16(v16h a, v16h b, v8f c) {
  // 8 args: (neg_a, A, neg_b, B, c_mod, C, reuse_a, reuse_b)
  return __builtin_amdgcn_wmma_f32_16x16x32_f16(false, a, false, b, (short)0, c,
                                                false, false);
}

// Issue one per-lane 16-byte async global->LDS copy (ASYNCcnt tracked).
__device__ inline void async_copy_b128(const float* gp, void* ldsp) {
  unsigned lds_off = (unsigned)(uintptr_t)ldsp;  // low 32 bits = LDS byte address
  asm volatile("global_load_async_to_lds_b128 %0, %1, off"
               :
               : "v"(lds_off), "v"(gp)
               : "memory");
}

__device__ inline void wait_asynccnt0() {
  asm volatile("s_wait_asynccnt 0x0" ::: "memory");
}

// ---- small utility kernels -------------------------------------------------

__global__ void fill_kernel(float* __restrict__ p, float v, size_t n) {
  size_t i = (size_t)blockIdx.x * blockDim.x + threadIdx.x;
  size_t stride = (size_t)gridDim.x * blockDim.x;
  for (; i < n; i += stride) p[i] = v;
}

__global__ void relu_kernel(float* __restrict__ p, size_t n) {
  size_t i = (size_t)blockIdx.x * blockDim.x + threadIdx.x;
  size_t stride = (size_t)gridDim.x * blockDim.x;
  for (; i < n; i += stride) p[i] = fmaxf(p[i], 0.0f);
}

// Pack a [K x N] row-major f32 weight into the WMMA B-fragment f16 layout.
__global__ void pack_w_kernel(const float* __restrict__ W, _Float16* __restrict__ P,
                              int K, int N) {
  int gid = blockIdx.x * blockDim.x + threadIdx.x;
  int total = K * N;
  if (gid >= total) return;
  int j    = gid & 15;
  int lane = (gid >> 4) & 31;
  int t    = gid >> 9;
  int nKt  = K >> 5;
  int kt   = t % nKt;
  int nt   = t / nKt;
  int base = (lane < 16) ? 0 : 8;
  int kl   = (j < 8) ? (base + j) : (16 + base + (j - 8));
  int k    = kt * 32 + kl;
  int n    = nt * 16 + (lane & 15);
  P[gid] = (_Float16)W[(size_t)k * N + n];
}

// Segment-sum scatter: agg[dst[e], :] += xsrc[src[e], :]; cnt[dst[e]] += 1.
__global__ void scatter_kernel(const float* __restrict__ xsrc, const int* __restrict__ src,
                               const int* __restrict__ dst, float* __restrict__ agg,
                               float* __restrict__ cnt, int nE, int C) {
  int chunks = C >> 2;
  long long gid = (long long)blockIdx.x * blockDim.x + threadIdx.x;
  long long total = (long long)nE * chunks;
  if (gid >= total) return;
  int e  = (int)(gid / chunks);
  int cq = (int)(gid - (long long)e * chunks);
  int s = src[e], d = dst[e];
  const float4 v = *(const float4*)(xsrc + (size_t)s * C + cq * 4);
  float* ap = agg + (size_t)d * C + cq * 4;
  atomicAdd(ap + 0, v.x);
  atomicAdd(ap + 1, v.y);
  atomicAdd(ap + 2, v.z);
  atomicAdd(ap + 3, v.w);
  if (cq == 0) atomicAdd(cnt + d, 1.0f);
}

// ---- fused SAGE tile kernel ------------------------------------------------
// Per block (128 thr = 4 waves): 16 dst rows x full N.
//   out = (agg/max(cnt,1)) @ Wl [if hasMean] + bias + xdst @ Wr
//   if doNorm: out /= max(||row||2, 1e-12)
//   outAcc[row, :] += relScale * out
__global__ void sage_tile_kernel(const float* __restrict__ agg, const float* __restrict__ cnt,
                                 const float* __restrict__ xdst,
                                 const _Float16* __restrict__ WlP,
                                 const _Float16* __restrict__ WrP,
                                 const float* __restrict__ bias,
                                 float* __restrict__ outAcc,
                                 int n_dst, int K, int N,
                                 float relScale, int hasMean, int doNorm) {
  __shared__ __align__(16) _Float16 shMean[16 * 256];
  __shared__ __align__(16) _Float16 shX[16 * 256];
  __shared__ __align__(16) float shXf[16 * 256];   // async f32 staging
  __shared__ __align__(16) float shMf[16 * 256];   // async f32 staging (mean path)
  __shared__ float shRowSq[4][16];
  __shared__ float shInv[16];

  const int tid  = threadIdx.x;
  const int row0 = blockIdx.x * 16;

  const bool fullTile = (row0 + 16) <= n_dst;
  if (fullTile) {
    // ---- async DMA of raw f32 rows into LDS (no VGPR round-trip) ----
    const int totalFloats = 16 * K;                // 2048 or 4096 floats
    for (int fo = tid * 4; fo < totalFloats; fo += TB * 4) {
      int m = fo / K;
      int k = fo - m * K;                          // K % 4 == 0: 16B stays in-row
      async_copy_b128(xdst + (size_t)(row0 + m) * K + k, &shXf[fo]);
      if (hasMean)
        async_copy_b128(agg + (size_t)(row0 + m) * K + k, &shMf[fo]);
    }
    wait_asynccnt0();
    __syncthreads();
    // ---- in-LDS f32 -> f16 conversion (+ mean division) ----
    for (int idx = tid; idx < totalFloats; idx += TB) {
      int m = idx / K;
      shX[idx] = (_Float16)shXf[idx];
      if (hasMean) {
        float c = cnt[row0 + m];
        shMean[idx] = (_Float16)(shMf[idx] / fmaxf(c, 1.0f));
      }
    }
  } else {
    // ---- ragged fallback: scalar staging with bounds guards ----
    for (int idx = tid; idx < 16 * K; idx += TB) {
      int m = idx / K;
      int k = idx - m * K;
      int r = row0 + m;
      float xv = (r < n_dst) ? xdst[(size_t)r * K + k] : 0.0f;
      shX[m * K + k] = (_Float16)xv;
      if (hasMean) {
        float c  = (r < n_dst) ? cnt[r] : 0.0f;
        float av = (r < n_dst) ? agg[(size_t)r * K + k] : 0.0f;
        shMean[m * K + k] = (_Float16)(av / fmaxf(c, 1.0f));
      }
    }
  }
  __syncthreads();

  const int lane = tid & 31;
  const int wave = tid >> 5;
  const int nTiles = N >> 4;        // 16 (N=256) or 8 (N=128)
  const int tilesPerWave = nTiles >> 2;
  const int nKt  = K >> 5;
  const int m_l  = lane & 15;
  const int hsel = (lane < 16) ? 0 : 8;

  v8f acc[4];
#pragma unroll
  for (int t = 0; t < 4; ++t) {
#pragma unroll
    for (int r = 0; r < 8; ++r) acc[t][r] = 0.0f;
  }

  for (int t = 0; t < tilesPerWave; ++t) {
    int nt = wave + (t << 2);
    // warm the B column panel for this wave (global_prefetch_b8)
    __builtin_prefetch(WrP + (((size_t)nt * nKt) << 9), 0, 1);
    v8f c = acc[t];
    if (hasMean) {
      for (int kt = 0; kt < nKt; ++kt) {
        v16h a = load_a_frag(shMean, m_l, kt << 5, K, hsel);
        v16h b = load_b_frag(WlP, nt, kt, nKt, lane);
        c = wmma_f16(a, b, c);
      }
    }
    for (int kt = 0; kt < nKt; ++kt) {
      v16h a = load_a_frag(shX, m_l, kt << 5, K, hsel);
      v16h b = load_b_frag(WrP, nt, kt, nKt, lane);
      c = wmma_f16(a, b, c);
    }
    float bv = bias[(nt << 4) + m_l];
#pragma unroll
    for (int r = 0; r < 8; ++r) c[r] += bv;
    acc[t] = c;
  }

  if (doNorm) {
    // per-row sum of squares: rows 0-7 live in lanes 0-15, rows 8-15 in 16-31
    float sq[8];
#pragma unroll
    for (int r = 0; r < 8; ++r) {
      float s = 0.0f;
      for (int t = 0; t < tilesPerWave; ++t) { float v = acc[t][r]; s += v * v; }
      sq[r] = s;
    }
#pragma unroll
    for (int off = 8; off > 0; off >>= 1) {
#pragma unroll
      for (int r = 0; r < 8; ++r) sq[r] += __shfl_xor(sq[r], off, 32);
    }
    if (m_l == 0) {
#pragma unroll
      for (int r = 0; r < 8; ++r) shRowSq[wave][hsel + r] = sq[r];
    }
    __syncthreads();
    if (tid < 16) {
      float s = shRowSq[0][tid] + shRowSq[1][tid] + shRowSq[2][tid] + shRowSq[3][tid];
      shInv[tid] = 1.0f / fmaxf(sqrtf(s), 1e-12f);
    }
    __syncthreads();
  }

  // scaled accumulate into destination buffer (RMW; launches are serialized)
  for (int t = 0; t < tilesPerWave; ++t) {
    int nt = wave + (t << 2);
    int n  = (nt << 4) + m_l;
#pragma unroll
    for (int r = 0; r < 8; ++r) {
      int m   = hsel + r;
      int row = row0 + m;
      if (row < n_dst) {
        float v = acc[t][r];
        if (doNorm) v *= shInv[m];
        size_t o = (size_t)row * N + n;
        outAcc[o] += relScale * v;
      }
    }
  }
}

// ---------------------------------------------------------------------------
// Host-side orchestration
// ---------------------------------------------------------------------------
extern "C" void kernel_launch(void* const* d_in, const int* in_sizes, int n_in,
                              void* d_out, int out_size, void* d_ws, size_t ws_size,
                              hipStream_t stream) {
  (void)n_in; (void)out_size; (void)ws_size;
  const int C_IN = 128, H = 256, OUT = 128;

  const float* x_user   = (const float*)d_in[0];
  const float* x_track  = (const float*)d_in[1];
  const float* x_artist = (const float*)d_in[2];
  const int nU = in_sizes[0] / C_IN;
  const int nT = in_sizes[1] / C_IN;
  const int nA = in_sizes[2] / C_IN;

  const int* esrc[4] = { (const int*)d_in[3], (const int*)d_in[5],
                         (const int*)d_in[7], (const int*)d_in[9] };
  const int* edst[4] = { (const int*)d_in[4], (const int*)d_in[6],
                         (const int*)d_in[8], (const int*)d_in[10] };
  const int  nE[4]   = { in_sizes[3], in_sizes[5], in_sizes[7], in_sizes[9] };

  // Params flattened jax-pytree style (sorted dict keys), starting at d_in[11]:
  //  conv1: by{Wl,Wr,bl} listens{...} rev_by{...} rev_listens{...}  -> 11..22
  //  conv2: same order                                               -> 23..34
  //  res  : artist{W,b} track{W,b} user{W,b}                         -> 35..40
  // Processing relation order: 0=listens 1=rev_listens 2=by 3=rev_by.
  const int pi1[4] = { 14, 20, 11, 17 };  // conv1 {Wl,Wr,bl} base index per relation
  const int pi2[4] = { 26, 32, 23, 29 };  // conv2 base index per relation
  const float* Wl1[4], *Wr1[4], *bl1[4], *Wl2[4], *Wr2[4], *bl2[4];
  for (int r = 0; r < 4; ++r) {
    Wl1[r] = (const float*)d_in[pi1[r] + 0];
    Wr1[r] = (const float*)d_in[pi1[r] + 1];
    bl1[r] = (const float*)d_in[pi1[r] + 2];
    Wl2[r] = (const float*)d_in[pi2[r] + 0];
    Wr2[r] = (const float*)d_in[pi2[r] + 1];
    bl2[r] = (const float*)d_in[pi2[r] + 2];
  }
  // res: order artist, track, user -> remap to {user,track,artist}
  const float* Wres[3] = { (const float*)d_in[39], (const float*)d_in[37], (const float*)d_in[35] };
  const float* bres[3] = { (const float*)d_in[40], (const float*)d_in[38], (const float*)d_in[36] };

  // --- workspace layout ---
  char* ws = (char*)d_ws;
  size_t off = 0;
  auto take = [&](size_t bytes) -> void* {
    void* p = ws + off;
    off = (off + bytes + 255) & ~(size_t)255;
    return p;
  };
  int maxNd = nT > nU ? nT : nU; if (nA > maxNd) maxNd = nA;
  float* agg  = (float*)take((size_t)maxNd * H * sizeof(float));   // reused, max n_dst x 256
  float* cntw = (float*)take((size_t)maxNd * sizeof(float));
  float* x1   = (float*)take((size_t)(nU + nT + nA) * H * sizeof(float));
  float* x1_user   = x1;
  float* x1_track  = x1 + (size_t)nU * H;
  float* x1_artist = x1_track + (size_t)nT * H;
  _Float16* packBase = (_Float16*)take((size_t)19 * 32768 * sizeof(_Float16));

  // --- 1. pack all weights to WMMA B-fragment f16 layout (each K*N == 32768) ---
  const float* wsrcs[19];
  int wK[19], wN[19];
  int wi = 0;
  for (int r = 0; r < 4; ++r) { wsrcs[wi] = Wl1[r]; wK[wi] = C_IN; wN[wi] = H; ++wi;
                                wsrcs[wi] = Wr1[r]; wK[wi] = C_IN; wN[wi] = H; ++wi; }
  for (int r = 0; r < 4; ++r) { wsrcs[wi] = Wl2[r]; wK[wi] = H; wN[wi] = OUT; ++wi;
                                wsrcs[wi] = Wr2[r]; wK[wi] = H; wN[wi] = OUT; ++wi; }
  for (int t = 0; t < 3; ++t) { wsrcs[wi] = Wres[t]; wK[wi] = H; wN[wi] = OUT; ++wi; }
  _Float16* pW[19];
  for (int i = 0; i < 19; ++i) {
    pW[i] = packBase + (size_t)i * 32768;
    pack_w_kernel<<<(32768 + 255) / 256, 256, 0, stream>>>(wsrcs[i], pW[i], wK[i], wN[i]);
  }
  _Float16 *pWl1[4], *pWr1[4], *pWl2[4], *pWr2[4], *pWres[3];
  for (int r = 0; r < 4; ++r) { pWl1[r] = pW[2 * r]; pWr1[r] = pW[2 * r + 1];
                                pWl2[r] = pW[8 + 2 * r]; pWr2[r] = pW[8 + 2 * r + 1]; }
  for (int t = 0; t < 3; ++t) pWres[t] = pW[16 + t];

  // relation metadata: src-type feature ptr, dst-type, n_dst, hetero-mean scale
  const float* xsrc1[4] = { x_user, x_track, x_track, x_artist };
  const float* xdst1[4] = { x_track, x_user, x_artist, x_track };
  float* h1src[4] = { x1_user, x1_track, x1_track, x1_artist };
  float* h1dst[4] = { x1_track, x1_user, x1_artist, x1_track };
  int    ndst[4]  = { nT, nU, nA, nT };
  float  scale[4] = { 0.5f, 1.0f, 1.0f, 0.5f };  // track has 2 incoming relations

  float* out_user   = (float*)d_out;
  float* out_track  = out_user + (size_t)nU * OUT;
  float* out_artist = out_track + (size_t)nT * OUT;
  float* outNT[3]   = { out_user, out_track, out_artist };
  float* h1NT[3]    = { x1_user, x1_track, x1_artist };
  int    nNT[3]     = { nU, nT, nA };
  float* outRel[4]  = { out_track, out_user, out_artist, out_track };

  // --- 2. layer 1 ---
  size_t x1n = (size_t)(nU + nT + nA) * H;
  fill_kernel<<<2048, 256, 0, stream>>>(x1, 0.0f, x1n);
  for (int r = 0; r < 4; ++r) {
    int nd = ndst[r];
    fill_kernel<<<2048, 256, 0, stream>>>(agg, 0.0f, (size_t)nd * C_IN);
    fill_kernel<<<256, 256, 0, stream>>>(cntw, 0.0f, (size_t)nd);
    long long work = (long long)nE[r] * (C_IN / 4);
    scatter_kernel<<<(int)((work + TB - 1) / TB), TB, 0, stream>>>(
        xsrc1[r], esrc[r], edst[r], agg, cntw, nE[r], C_IN);
    sage_tile_kernel<<<(nd + 15) / 16, TB, 0, stream>>>(
        agg, cntw, xdst1[r], pWl1[r], pWr1[r], bl1[r], h1dst[r],
        nd, C_IN, H, scale[r], 1, 1);
  }
  relu_kernel<<<2048, 256, 0, stream>>>(x1, x1n);

  // --- 3. layer 2: residual GEMMs + relation SAGE, all accumulated into d_out ---
  fill_kernel<<<2048, 256, 0, stream>>>((float*)d_out,
                                        0.0f, (size_t)(nU + nT + nA) * OUT);
  for (int t = 0; t < 3; ++t) {
    sage_tile_kernel<<<(nNT[t] + 15) / 16, TB, 0, stream>>>(
        nullptr, nullptr, h1NT[t], nullptr, pWres[t], bres[t], outNT[t],
        nNT[t], H, OUT, 1.0f, 0, 0);
  }
  for (int r = 0; r < 4; ++r) {
    int nd = ndst[r];
    fill_kernel<<<2048, 256, 0, stream>>>(agg, 0.0f, (size_t)nd * H);
    fill_kernel<<<256, 256, 0, stream>>>(cntw, 0.0f, (size_t)nd);
    long long work = (long long)nE[r] * (H / 4);
    scatter_kernel<<<(int)((work + TB - 1) / TB), TB, 0, stream>>>(
        h1src[r], esrc[r], edst[r], agg, cntw, nE[r], H);
    sage_tile_kernel<<<(nd + 15) / 16, TB, 0, stream>>>(
        agg, cntw, h1dst[r], pWl2[r], pWr2[r], bl2[r], outRel[r],
        nd, H, OUT, scale[r], 1, 1);
  }
}
